// ResGCN_58317065945288
// MI455X (gfx1250) — compile-verified, matrix operand order checked by reading, not run on hardware
//
#include <hip/hip_runtime.h>
#include <hip/hip_bf16.h>
#include <stdint.h>

// ---------------------------------------------------------------------------
// Types for WMMA
// ---------------------------------------------------------------------------
typedef __attribute__((ext_vector_type(16))) __bf16 v16bf;
typedef __attribute__((ext_vector_type(8)))  __bf16 v8bf;
typedef __attribute__((ext_vector_type(8)))  float  v8f;

#define DK   512          // feature dim (K == N for the big GEMMs)
#define FEPS 1e-12f

__device__ __forceinline__ unsigned short f2bf(float f) {
  unsigned int u = __float_as_uint(f);
  u += 0x7FFFu + ((u >> 16) & 1u);          // round-to-nearest-even
  return (unsigned short)(u >> 16);
}
__device__ __forceinline__ unsigned int pk2(float a, float b) {
  return (unsigned int)f2bf(a) | ((unsigned int)f2bf(b) << 16);
}

// ---------------------------------------------------------------------------
// W (KxN f32, row-major) -> Wt (NxK bf16, row-major)  [tiny: 512x512]
// ---------------------------------------------------------------------------
__global__ void transpose_w_bf16(const float* __restrict__ W,
                                 unsigned short* __restrict__ Wt) {
  int idx = blockIdx.x * blockDim.x + threadIdx.x;   // n*512 + k
  int n = idx >> 9, k = idx & 511;
  Wt[idx] = f2bf(W[k * DK + n]);
}

// ---------------------------------------------------------------------------
// GEMM: C[M x 512] = A[M x 512] (f32 -> bf16) @ Wt^T  via WMMA bf16.
// One block owns 128 rows and ALL 512 output cols:
//   - whole 128x512 bf16 A tile staged in LDS once (130 KB of the 320 KB WGP LDS)
//   - 4 N-slices of 128 computed back-to-back, streaming Bt (L2-resident)
//   - explicit double-buffered sB: one barrier per K-step
// => A is read from HBM exactly once per GEMM.
// 256 threads = 8 waves (2x4); per-wave 4x2 WMMA 16x16 tiles per slice.
// ---------------------------------------------------------------------------
#define MT   128
#define NT   128
#define KT   32
#define LDB  40     // sB row stride (ushorts): 32 + 8 pad  -> 80B
#define LDA2 520    // sA row stride (ushorts): 512 + 8 pad -> 1040B (16B aligned)

__device__ __forceinline__
void load_B(const unsigned short* __restrict__ Bt, int n0, int kc,
            int br, int bk, uint4 rb[2]) {
#pragma unroll
  for (int g = 0; g < 2; g++)
    rb[g] = *(const uint4*)(Bt + (size_t)(n0 + g * 64 + br) * DK + kc + bk);
}

__global__ __launch_bounds__(256)
void gemm_bf16_wmma(const float* __restrict__ A,
                    const unsigned short* __restrict__ Bt,  // [512][512] bf16, N-major
                    float* __restrict__ C, int M) {
  __shared__ unsigned short sA[MT * LDA2];     // 133,120 B
  __shared__ unsigned short sB[2][NT * LDB];   // 2 x 10,240 B

  const int tid  = threadIdx.x;
  const int lane = tid & 31;
  const int wid  = tid >> 5;
  const int wr   = wid >> 2;            // 0..1
  const int wc   = wid & 3;             // 0..3
  const int m0   = blockIdx.x * MT;
  const bool full = (m0 + MT) <= M;     // scalar: all rows in range?

  // global->LDS load mapping (8 threads cover a 128B row segment)
  const int ar = tid >> 3;              // 0..31 (A row group base)
  const int ak = (tid & 7) * 4;         // k offset within 32
  const int br = tid >> 2;              // 0..63 (B row group base)
  const int bk = (tid & 3) * 8;

  // fragment lane mapping (ISA 16-bit A-matrix 16x32 layout)
  const int lm = lane & 15;
  const int kl = (lane >> 4) << 3;      // 0 or 8 (ushort offset)

  // ---- stage the full 128x512 A tile into LDS (f32 -> bf16) ----
#pragma unroll 4
  for (int kc = 0; kc < DK; kc += KT) {
#pragma unroll
    for (int g = 0; g < 4; g++) {
      int r  = g * 32 + ar;
      int gm = m0 + r;
      gm = (gm < M) ? gm : (M - 1);     // clamp: rows >= M never stored
      float4 f = *(const float4*)(A + (size_t)gm * DK + kc + ak);
      uint2 p;
      p.x = pk2(f.x, f.y);
      p.y = pk2(f.z, f.w);
      *(uint2*)&sA[r * LDA2 + kc + ak] = p;
    }
  }

  // ---- prime sB buffer 0 ----
  uint4 rb[2];
  load_B(Bt, 0, 0, br, bk, rb);
#pragma unroll
  for (int g = 0; g < 2; g++)
    *(uint4*)&sB[0][(g * 64 + br) * LDB + bk] = rb[g];

  __syncthreads();                      // A tile + sB[0] ready

  v8f acc[4][2];

  // flattened (n-slice, k-step) loop: 4 * 16 = 64 steps, one barrier per step
  for (int s = 0; s < 64; s++) {
    const int ns  = s >> 4;
    const int kc  = (s & 15) * KT;
    const int n0  = ns * NT;
    const int cur = s & 1;

    if ((s & 15) == 0) {
#pragma unroll
      for (int i = 0; i < 4; i++)
#pragma unroll
        for (int j = 0; j < 2; j++) { v8f z = {}; acc[i][j] = z; }
    }

    // prefetch next B chunk straight into the inactive sB half
    if (s + 1 < 64) {
      const int s1 = s + 1;
      load_B(Bt, (s1 >> 4) * NT, (s1 & 15) * KT, br, bk, rb);
#pragma unroll
      for (int g = 0; g < 2; g++)
        *(uint4*)&sB[s1 & 1][(g * 64 + br) * LDB + bk] = rb[g];
    }

    v16bf af[4], bfr[2];
#pragma unroll
    for (int i = 0; i < 4; i++) {
      const unsigned short* p = &sA[(wr * 64 + i * 16 + lm) * LDA2 + kc + kl];
      v8bf lo = *(const v8bf*)p;
      v8bf hi = *(const v8bf*)(p + 16);
      af[i] = __builtin_shufflevector(lo, hi, 0,1,2,3,4,5,6,7,8,9,10,11,12,13,14,15);
    }
#pragma unroll
    for (int j = 0; j < 2; j++) {
      const unsigned short* p = &sB[cur][(wc * 32 + j * 16 + lm) * LDB + kl];
      v8bf lo = *(const v8bf*)p;
      v8bf hi = *(const v8bf*)(p + 16);
      bfr[j] = __builtin_shufflevector(lo, hi, 0,1,2,3,4,5,6,7,8,9,10,11,12,13,14,15);
    }
#pragma unroll
    for (int i = 0; i < 4; i++)
#pragma unroll
      for (int j = 0; j < 2; j++)
        acc[i][j] = __builtin_amdgcn_wmma_f32_16x16x32_bf16(
            false, af[i], false, bfr[j], (short)0, acc[i][j], false, false);

    if ((s & 15) == 15) {
      // C layout: N = lane%16, M = vgpr + 8*(lane>=16)
      const int cn = n0 + wc * 32 + lm;
      const int mb = m0 + wr * 64 + ((lane >> 4) << 3);
      if (full) {
        // common path: straight-line stores, no per-element predication
#pragma unroll
        for (int i = 0; i < 4; i++) {
#pragma unroll
          for (int v = 0; v < 8; v++) {
            float* crow = C + (size_t)(mb + i * 16 + v) * DK + cn;
            crow[0]  = acc[i][0][v];
            crow[16] = acc[i][1][v];
          }
        }
      } else {
#pragma unroll
        for (int i = 0; i < 4; i++) {
#pragma unroll
          for (int v = 0; v < 8; v++) {
            int m = mb + i * 16 + v;
            if (m < M) {
              float* crow = C + (size_t)m * DK + cn;
              crow[0]  = acc[i][0][v];
              crow[16] = acc[i][1][v];
            }
          }
        }
      }
    }

    __syncthreads();   // next-buffer writes visible; current buffer free for s+2
  }
}

// ---------------------------------------------------------------------------
// Degree / dinv
// ---------------------------------------------------------------------------
__global__ void deg_init(float* deg, int n) {
  int i = blockIdx.x * blockDim.x + threadIdx.x;
  if (i < n) deg[i] = 1.0f;                      // self-loop weight
}
__global__ void deg_scatter(float* deg, const int* __restrict__ col,
                            const float* __restrict__ w, int e) {
  int i = blockIdx.x * blockDim.x + threadIdx.x;
  if (i < e) atomicAdd(&deg[col[i]], w[i]);
}
__global__ void deg_to_dinv(float* d, int n) {
  int i = blockIdx.x * blockDim.x + threadIdx.x;
  if (i < n) {
    float v = d[i];
    d[i] = (v > 0.f) ? rsqrtf(fmaxf(v, FEPS)) : 0.f;
  }
}

// ---------------------------------------------------------------------------
// conv init: out[i][c] = bias[c] + dinv[i]^2 * mm[i][c]   (self-loop term)
// ---------------------------------------------------------------------------
__global__ void conv_init(float* __restrict__ out, const float* __restrict__ mm,
                          const float* __restrict__ dinv,
                          const float* __restrict__ bias, int n) {
  int idx = blockIdx.x * blockDim.x + threadIdx.x;       // over n*128 float4s
  int total = n * (DK / 4);
  if (idx >= total) return;
  int i  = idx >> 7;           // node
  int c4 = idx & 127;          // float4 column
  float di = dinv[i];
  float s = di * di;
  float4 m = ((const float4*)(mm + (size_t)i * DK))[c4];
  float4 b = ((const float4*)bias)[c4];
  float4 o;
  o.x = b.x + s * m.x; o.y = b.y + s * m.y;
  o.z = b.z + s * m.z; o.w = b.w + s * m.w;
  ((float4*)(out + (size_t)i * DK))[c4] = o;
}

// ---------------------------------------------------------------------------
// edge scatter: out[col] += dinv[row]*w*dinv[col] * mm[row]   (one block/edge)
// ---------------------------------------------------------------------------
__global__ __launch_bounds__(128)
void conv_scatter(float* __restrict__ out, const float* __restrict__ mm,
                  const float* __restrict__ dinv,
                  const int* __restrict__ row, const int* __restrict__ col,
                  const float* __restrict__ w) {
  int e = blockIdx.x;
  int r = row[e], c = col[e];
  float nrm = dinv[r] * w[e] * dinv[c];
  int t = threadIdx.x;
  float4 v = ((const float4*)(mm + (size_t)r * DK))[t];
  float* dst = out + (size_t)c * DK + t * 4;
  atomicAdd(dst + 0, nrm * v.x);
  atomicAdd(dst + 1, nrm * v.y);
  atomicAdd(dst + 2, nrm * v.z);
  atomicAdd(dst + 3, nrm * v.w);
}

// ---------------------------------------------------------------------------
// block reduce helper (128 threads = 4 waves, wave32)
// ---------------------------------------------------------------------------
__device__ __forceinline__ float breduce128(float v, float* sm, int tid) {
#pragma unroll
  for (int o = 16; o > 0; o >>= 1) v += __shfl_down(v, o, 32);
  if ((tid & 31) == 0) sm[tid >> 5] = v;
  __syncthreads();
  if (tid == 0) sm[0] = sm[0] + sm[1] + sm[2] + sm[3];
  __syncthreads();
  float r = sm[0];
  __syncthreads();
  return r;
}

// ---------------------------------------------------------------------------
// finalize: h = relu(conv); h /= max(||h||,eps); h = (h + xin) * 0.5  (in place)
// ---------------------------------------------------------------------------
__global__ __launch_bounds__(128)
void finalize_block(float* __restrict__ conv, const float* __restrict__ xin) {
  __shared__ float sm[8];
  int i = blockIdx.x;
  int t = threadIdx.x;
  float4 v = ((const float4*)(conv + (size_t)i * DK))[t];
  v.x = fmaxf(v.x, 0.f); v.y = fmaxf(v.y, 0.f);
  v.z = fmaxf(v.z, 0.f); v.w = fmaxf(v.w, 0.f);
  float ss = v.x * v.x + v.y * v.y + v.z * v.z + v.w * v.w;
  float tot = breduce128(ss, sm, t);
  float inv = 1.0f / fmaxf(sqrtf(tot), FEPS);
  float4 x = ((const float4*)(xin + (size_t)i * DK))[t];
  float4 o;
  o.x = (v.x * inv + x.x) * 0.5f; o.y = (v.y * inv + x.y) * 0.5f;
  o.z = (v.z * inv + x.z) * 0.5f; o.w = (v.w * inv + x.w) * 0.5f;
  ((float4*)(conv + (size_t)i * DK))[t] = o;
}

// ---------------------------------------------------------------------------
// final projection (512 -> 2): mmf[i] = H2[i] @ Wf ; conv3 init with bias+selfloop
// ---------------------------------------------------------------------------
__global__ __launch_bounds__(128)
void final_proj(const float* __restrict__ H, const float* __restrict__ Wf,
                const float* __restrict__ bf, const float* __restrict__ dinv,
                float* __restrict__ mmf, float* __restrict__ cv3) {
  __shared__ float sm[8];
  int i = blockIdx.x;
  int t = threadIdx.x;
  float4 h  = ((const float4*)(H + (size_t)i * DK))[t];
  const float4* wf4 = (const float4*)Wf;         // Wf[k][j], j fastest
  float4 wa = wf4[t * 2 + 0];                    // k=4t..4t+1
  float4 wb = wf4[t * 2 + 1];                    // k=4t+2..4t+3
  float s0 = h.x * wa.x + h.y * wa.z + h.z * wb.x + h.w * wb.z;
  float s1 = h.x * wa.y + h.y * wa.w + h.z * wb.y + h.w * wb.w;
  float t0 = breduce128(s0, sm, t);
  float t1 = breduce128(s1, sm, t);
  if (t == 0) {
    float di = dinv[i], s = di * di;
    mmf[i * 2 + 0] = t0;
    mmf[i * 2 + 1] = t1;
    cv3[i * 2 + 0] = bf[0] + s * t0;
    cv3[i * 2 + 1] = bf[1] + s * t1;
  }
}

__global__ void scatter2(float* __restrict__ cv3, const float* __restrict__ mmf,
                         const float* __restrict__ dinv,
                         const int* __restrict__ row, const int* __restrict__ col,
                         const float* __restrict__ w, int e) {
  int i = blockIdx.x * blockDim.x + threadIdx.x;
  if (i >= e) return;
  int r = row[i], c = col[i];
  float nrm = dinv[r] * w[i] * dinv[c];
  atomicAdd(&cv3[c * 2 + 0], nrm * mmf[r * 2 + 0]);
  atomicAdd(&cv3[c * 2 + 1], nrm * mmf[r * 2 + 1]);
}

__global__ void log_softmax2(float* __restrict__ cv3, int n) {
  int i = blockIdx.x * blockDim.x + threadIdx.x;
  if (i >= n) return;
  float a = cv3[i * 2], b = cv3[i * 2 + 1];
  float m = fmaxf(a, b);
  float lse = m + logf(expf(a - m) + expf(b - m));
  cv3[i * 2]     = a - lse;
  cv3[i * 2 + 1] = b - lse;
}

__global__ void pred_dot(float* __restrict__ out, const float* __restrict__ ls,
                         const int* __restrict__ pu, const int* __restrict__ pv,
                         int p) {
  int i = blockIdx.x * blockDim.x + threadIdx.x;
  if (i >= p) return;
  int u = pu[i], v = pv[i];
  out[i * 2 + 0] = -(ls[u * 2 + 0] * ls[v * 2 + 0]);
  out[i * 2 + 1] = -(ls[u * 2 + 1] * ls[v * 2 + 1]);
}

// ---------------------------------------------------------------------------
// host launcher
// ---------------------------------------------------------------------------
static inline size_t alignup(size_t v) { return (v + 255) & ~(size_t)255; }

extern "C" void kernel_launch(void* const* d_in, const int* in_sizes, int n_in,
                              void* d_out, int out_size, void* d_ws, size_t ws_size,
                              hipStream_t stream) {
  const float* x   = (const float*)d_in[0];
  const int*   ei  = (const int*)d_in[1];
  const float* ew  = (const float*)d_in[2];
  const int*   pe  = (const int*)d_in[3];
  const float* W1  = (const float*)d_in[4];
  const float* b1  = (const float*)d_in[5];
  const float* W2  = (const float*)d_in[6];
  const float* b2  = (const float*)d_in[7];
  const float* Wf  = (const float*)d_in[8];
  const float* bf  = (const float*)d_in[9];
  float* out = (float*)d_out;

  const int N = in_sizes[0] / DK;
  const int E = in_sizes[2];
  const int P = in_sizes[3] / 2;
  const int* row = ei;
  const int* col = ei + E;
  const int* pu  = pe;
  const int* pv  = pe + P;

  char* ws = (char*)d_ws;
  size_t off = 0;
  unsigned short* Wt = (unsigned short*)(ws + off); off += alignup((size_t)DK * DK * 2);
  float* dinv = (float*)(ws + off); off += alignup((size_t)N * 4);
  float* mm   = (float*)(ws + off); off += alignup((size_t)N * DK * 4);
  float* bufA = (float*)(ws + off); off += alignup((size_t)N * DK * 4);
  float* bufB = (float*)(ws + off); off += alignup((size_t)N * DK * 4);
  float* mmf  = (float*)(ws + off); off += alignup((size_t)N * 2 * 4);
  float* cv3  = (float*)(ws + off); off += alignup((size_t)N * 2 * 4);
  (void)ws_size; (void)n_in; (void)out_size;

  dim3 gemm_grid((N + MT - 1) / MT);     // each block covers all 512 output cols
  int nblk  = (N + 255) / 256;
  int eblk  = (E + 255) / 256;
  int pblk  = (P + 255) / 256;
  int ciblk = (N * (DK / 4) + 255) / 256;

  // degree -> dinv
  deg_init<<<nblk, 256, 0, stream>>>(dinv, N);
  deg_scatter<<<eblk, 256, 0, stream>>>(dinv, col, ew, E);
  deg_to_dinv<<<nblk, 256, 0, stream>>>(dinv, N);

  // ---- block 1 ----
  transpose_w_bf16<<<(DK * DK) / 256, 256, 0, stream>>>(W1, Wt);
  gemm_bf16_wmma<<<gemm_grid, 256, 0, stream>>>(x, Wt, mm, N);
  conv_init<<<ciblk, 256, 0, stream>>>(bufA, mm, dinv, b1, N);
  conv_scatter<<<E, 128, 0, stream>>>(bufA, mm, dinv, row, col, ew);
  finalize_block<<<N, 128, 0, stream>>>(bufA, x);        // bufA = H1

  // ---- block 2 ----
  transpose_w_bf16<<<(DK * DK) / 256, 256, 0, stream>>>(W2, Wt);
  gemm_bf16_wmma<<<gemm_grid, 256, 0, stream>>>(bufA, Wt, mm, N);
  conv_init<<<ciblk, 256, 0, stream>>>(bufB, mm, dinv, b2, N);
  conv_scatter<<<E, 128, 0, stream>>>(bufB, mm, dinv, row, col, ew);
  finalize_block<<<N, 128, 0, stream>>>(bufB, bufA);     // bufB = H2

  // ---- final conv (512 -> 2) + log_softmax + predictor ----
  final_proj<<<N, 128, 0, stream>>>(bufB, Wf, bf, dinv, mmf, cv3);
  scatter2<<<eblk, 256, 0, stream>>>(cv3, mmf, dinv, row, col, ew, E);
  log_softmax2<<<nblk, 256, 0, stream>>>(cv3, N);
  pred_dot<<<pblk, 256, 0, stream>>>(out, cv3, pu, pv, P);
}
